// Encoding_layer_63986422775838
// MI455X (gfx1250) — matrix operand order repaired and checked by reading, not compiled, not run on hardware
//
#include <hip/hip_runtime.h>
#include <math.h>

// Problem constants (match reference): B=32, T=512, D=H=512, gates G=4H=2048.
#define Bz 32
#define Tz 512
#define Dz 512
#define Hz 512
#define Gz 2048
#define BTz (Bz * Tz)
#define NBLK 64  // persistent scan blocks: 2 m-tiles * 32 n-tiles

typedef __attribute__((ext_vector_type(16))) __bf16 v16bf;
typedef __attribute__((ext_vector_type(8))) float v8f;

__device__ __forceinline__ float sigm(float x) { return 1.0f / (1.0f + expf(-x)); }

#define WMMA_BF16(a, b, c) \
  __builtin_amdgcn_wmma_f32_16x16x32_bf16(false, (a), false, (b), (short)0, (c), false, false)

// ---------------------------------------------------------------------------
// WMMA fragment loaders (CDNA5 16x16x32 bf16 layouts, ISA 7.12.2)
// A: 16x32 (MxK) bf16, row-major, lda elements.
//   lane l: row = l&15 ; elems 0..7  = K[k0+base .. +7]   (base = l<16?0:8)
//           elems 8..15 = K[k0+base+16 .. +23]  -> two contiguous b128 loads
__device__ __forceinline__ v16bf load_a_frag(const __bf16* __restrict__ A, int lda,
                                             int row0, int k0) {
  const int lane = threadIdx.x & 31;
  const __bf16* p = A + (size_t)(row0 + (lane & 15)) * lda + k0 + ((lane < 16) ? 0 : 8);
  v16bf a;
#pragma unroll
  for (int e = 0; e < 8; ++e) a[e] = p[e];
#pragma unroll
  for (int e = 0; e < 8; ++e) a[8 + e] = p[16 + e];
  return a;
}

// B: 32x16 (KxN) bf16 sourced from pre-transposed weights WT[N][K] (ldb = K).
//   lane l: col = l&15 ; elems 0..7 = K[k0+base .. +7], elems 8..15 = K[k0+base+16 .. +23] (base = l<16?0:8)
__device__ __forceinline__ v16bf load_b_frag(const __bf16* __restrict__ WT, int ldb,
                                             int n0, int k0) {
  const int lane = threadIdx.x & 31;
  const __bf16* p = WT + (size_t)(n0 + (lane & 15)) * ldb + k0 + ((lane < 16) ? 0 : 8);
  v16bf b;
#pragma unroll
  for (int e = 0; e < 8; ++e) b[e] = p[e];
#pragma unroll
  for (int e = 0; e < 8; ++e) b[8 + e] = p[16 + e];
  return b;
}

// ---------------------------------------------------------------------------
// Time-parallel GEMM: C[M,N] = A[M,K](bf16) * WT[N,K](bf16) + bias[N]
// One wave per 16x64 output strip: A-fragment reused across 4 WMMAs.
__global__ __launch_bounds__(256) void k_gemm4(const __bf16* __restrict__ A,
                                               const __bf16* __restrict__ WT,
                                               const float* __restrict__ bias,
                                               float* __restrict__ C,
                                               int M, int N, int K) {
  const int wave = (blockIdx.x * blockDim.x + threadIdx.x) >> 5;
  const int ng = N >> 6;  // groups of four 16-wide n-tiles
  const int total = (M >> 4) * ng;
  if (wave >= total) return;  // wave-uniform branch, EXEC stays all-ones for WMMA
  const int mt = wave / ng;
  const int n0 = (wave % ng) << 6;
  const int lane = threadIdx.x & 31;

  v8f acc0 = {}, acc1 = {}, acc2 = {}, acc3 = {};
  const __bf16* arow = A + (size_t)(mt * 16 + (lane & 15)) * K + ((lane < 16) ? 0 : 8);
  for (int k = 0; k < K; k += 32) {
    if (k + 32 < K) __builtin_prefetch(arow + k + 32, 0, 3);  // global_prefetch_b8
    v16bf a = load_a_frag(A, K, mt * 16, k);
    v16bf b0 = load_b_frag(WT, K, n0, k);
    v16bf b1 = load_b_frag(WT, K, n0 + 16, k);
    v16bf b2 = load_b_frag(WT, K, n0 + 32, k);
    v16bf b3 = load_b_frag(WT, K, n0 + 48, k);
    acc0 = WMMA_BF16(a, b0, acc0);
    acc1 = WMMA_BF16(a, b1, acc1);
    acc2 = WMMA_BF16(a, b2, acc2);
    acc3 = WMMA_BF16(a, b3, acc3);
  }
  // C/D layout: VGPR r -> row r + (lane<16?0:8), col = lane&15
  const int mrow = mt * 16 + ((lane < 16) ? 0 : 8);
  const int ncol = lane & 15;
#pragma unroll
  for (int q = 0; q < 4; ++q) {
    const v8f acc = (q == 0) ? acc0 : (q == 1) ? acc1 : (q == 2) ? acc2 : acc3;
    const int n = n0 + q * 16 + ncol;
    const float bv = bias ? bias[n] : 0.0f;
#pragma unroll
    for (int r = 0; r < 8; ++r) C[(size_t)(mrow + r) * N + n] = acc[r] + bv;
  }
}

// ---------------------------------------------------------------------------
// Persistent LSTM scan. 64 blocks x 128 threads (4 waves).
// Block = one 16(batch) x 16(hidden) cell; wave g computes gate g tile.
// Wh fragments for this wave's gate tile are PRELOADED into registers
// (16 x v16bf = 128 VGPRs) so the per-step inner loop is WMMA + A-loads only.
__global__ __launch_bounds__(128) void k_lstm_scan(const float* __restrict__ xg,
                                                   const __bf16* __restrict__ WhT,
                                                   const int* __restrict__ length,
                                                   float* __restrict__ y,
                                                   __bf16* __restrict__ hbuf,
                                                   float* __restrict__ cstate,
                                                   unsigned* __restrict__ cnt) {
  __shared__ float gates[4][16][16];

  const int blk = blockIdx.x;  // 0..63
  const int mt = blk >> 5;     // 0..1 (batch tile)
  const int nt = blk & 31;     // 0..31 (hidden tile)
  const int tid = threadIdx.x;
  const int wv = tid >> 5;     // gate index: 0=i 1=j 2=f 3=o
  const int lane = tid & 31;
  const int mrow0 = mt * 16;
  const int gcol = wv * Hz + nt * 16 + (lane & 15);

  // Register-resident recurrent weights: full K=512 slice of this gate tile.
  v16bf bw[16];
#pragma unroll
  for (int kk = 0; kk < 16; ++kk)
    bw[kk] = load_b_frag(WhT, Hz, wv * Hz + nt * 16, kk * 32);

  // Per-lane xg base offsets for the 8 accumulator rows (row m fixed per r).
  size_t xgoff[8];
#pragma unroll
  for (int r = 0; r < 8; ++r) {
    const int m = mrow0 + r + ((lane < 16) ? 0 : 8);
    xgoff[r] = (size_t)m * Tz * Gz + gcol;
  }

  // per-thread elementwise assignment: rows rloc and rloc+8, column colL
  const int rloc = tid >> 4;  // 0..7
  const int colL = tid & 15;
  const int hcol = nt * 16 + colL;
  const int batchA = mrow0 + rloc;
  const int batchB = mrow0 + rloc + 8;
  const int lenA = length[batchA];
  const int lenB = length[batchB];

  for (int t = 0; t < Tz; ++t) {
    const __bf16* hc = hbuf + (size_t)(t & 1) * (Bz * Hz);
    __bf16* hn = hbuf + (size_t)((t + 1) & 1) * (Bz * Hz);

    // Gate pre-activation: acc = h_t @ Wh  (16 WMMAs, weights already in VGPRs)
    v8f acc = {};
#pragma unroll
    for (int kk = 0; kk < 16; ++kk) {
      v16bf a = load_a_frag(hc, Hz, mrow0, kk * 32);
      acc = WMMA_BF16(a, bw[kk], acc);
    }
    // add xg[t] and park in LDS for the cross-gate cell update
#pragma unroll
    for (int r = 0; r < 8; ++r) {
      const int mloc = r + ((lane < 16) ? 0 : 8);
      gates[wv][mloc][lane & 15] = acc[r] + xg[xgoff[r] + (size_t)t * Gz];
    }
    __syncthreads();

    // Cell update (TF gate order i, j, f, o; forget_bias = 1.0)
#pragma unroll
    for (int half = 0; half < 2; ++half) {
      const int mloc = rloc + half * 8;
      const int batch = half ? batchB : batchA;
      const int len = half ? lenB : lenA;
      const float gi = gates[0][mloc][colL];
      const float gj = gates[1][mloc][colL];
      const float gf = gates[2][mloc][colL];
      const float go = gates[3][mloc][colL];
      const size_t sidx = (size_t)batch * Hz + hcol;
      const float c_old = cstate[sidx];
      const float c_new = sigm(gf + 1.0f) * c_old + sigm(gi) * tanhf(gj);
      const float h_new = sigm(go) * tanhf(c_new);
      const bool live = t < len;
      const float h_old = (float)hc[sidx];
      cstate[sidx] = live ? c_new : c_old;
      hn[sidx] = (__bf16)(live ? h_new : h_old);
      y[((size_t)batch * Tz + t) * Hz + hcol] = live ? h_new : 0.0f;
    }
    __syncthreads();

    // Grid-wide step barrier (monotone counter; target is deterministic in t)
    __threadfence();
    if (tid == 0) {
      __hip_atomic_fetch_add(cnt, 1u, __ATOMIC_ACQ_REL, __HIP_MEMORY_SCOPE_AGENT);
      const unsigned target = (unsigned)(t + 1) * NBLK;
      while (__hip_atomic_load(cnt, __ATOMIC_ACQUIRE, __HIP_MEMORY_SCOPE_AGENT) < target) {
        __builtin_amdgcn_s_sleep(1);
      }
    }
    __syncthreads();
  }
}

// ---------------------------------------------------------------------------
// Small elementwise helpers
__global__ void k_scan_init(unsigned* cnt, float* cst, __bf16* hb) {
  const int idx = blockIdx.x * blockDim.x + threadIdx.x;
  if (idx == 0) *cnt = 0u;
  if (idx < Bz * Hz) {
    cst[idx] = 0.0f;
    hb[idx] = (__bf16)0.0f;
    hb[Bz * Hz + idx] = (__bf16)0.0f;
  }
}

// WT[n*K + k] = bf16(W[k*N + n])   (W is [K,N] f32)
__global__ void k_transpose_bf16(const float* __restrict__ W, __bf16* __restrict__ WT,
                                 int K, int N) {
  const int idx = blockIdx.x * blockDim.x + threadIdx.x;
  if (idx >= K * N) return;
  const int n = idx / K;
  const int k = idx % K;
  WT[idx] = (__bf16)W[(size_t)k * N + n];
}

__global__ void k_f32_to_bf16(const float* __restrict__ in, __bf16* __restrict__ out,
                              int n) {
  const int idx = blockIdx.x * blockDim.x + threadIdx.x;
  if (idx < n) out[idx] = (__bf16)in[idx];
}

// reverse_sequence over time axis with per-batch length, fused f32->bf16
__global__ void k_reverse_bf16(const float* __restrict__ in, const int* __restrict__ len,
                               __bf16* __restrict__ out) {
  const int idx = blockIdx.x * blockDim.x + threadIdx.x;
  if (idx >= BTz * Hz) return;
  const int h = idx % Hz;
  const int r = idx / Hz;
  const int t = r % Tz;
  const int b = r / Tz;
  const int L = len[b];
  const int ts = (t < L) ? (L - 1 - t) : t;
  out[idx] = (__bf16)in[((size_t)b * Tz + ts) * Hz + h];
}

__global__ void k_reverse_f32(const float* __restrict__ in, const int* __restrict__ len,
                              float* __restrict__ out) {
  const int idx = blockIdx.x * blockDim.x + threadIdx.x;
  if (idx >= BTz * Hz) return;
  const int h = idx % Hz;
  const int r = idx / Hz;
  const int t = r % Tz;
  const int b = r / Tz;
  const int L = len[b];
  const int ts = (t < L) ? (L - 1 - t) : t;
  out[idx] = in[((size_t)b * Tz + ts) * Hz + h];
}

// cc[r, 0:H] = bf16(prev_h)  (shift of yout, zero at t=0);  cc[r, H:2H] = bf16(x)
__global__ void k_concat(const float* __restrict__ yout, const float* __restrict__ x,
                         __bf16* __restrict__ cc) {
  const int idx = blockIdx.x * blockDim.x + threadIdx.x;
  if (idx >= BTz * Hz) return;
  const int h = idx % Hz;
  const int r = idx / Hz;
  const int t = r % Tz;
  const float pv = (t > 0) ? yout[(size_t)(r - 1) * Hz + h] : 0.0f;
  cc[(size_t)r * (2 * Hz) + h] = (__bf16)pv;
  cc[(size_t)r * (2 * Hz) + Hz + h] = (__bf16)x[idx];
}

// out = y * sigmoid(tg) + cl * (1 - sigmoid(tg))
__global__ void k_combine(const float* __restrict__ yv, const float* __restrict__ tg,
                          const float* __restrict__ cl, float* __restrict__ out, int n) {
  const int idx = blockIdx.x * blockDim.x + threadIdx.x;
  if (idx >= n) return;
  const float s = sigm(tg[idx]);
  out[idx] = yv[idx] * s + cl[idx] * (1.0f - s);
}

// ---------------------------------------------------------------------------
extern "C" void kernel_launch(void* const* d_in, const int* in_sizes, int n_in,
                              void* d_out, int out_size, void* d_ws, size_t ws_size,
                              hipStream_t stream) {
  (void)in_sizes; (void)n_in; (void)out_size; (void)ws_size;
  const float* inputs = (const float*)d_in[0];
  const int* length = (const int*)d_in[1];
  const float* Wx1 = (const float*)d_in[3];
  const float* Wh1 = (const float*)d_in[4];
  const float* b1  = (const float*)d_in[5];
  const float* Wx2 = (const float*)d_in[6];
  const float* Wh2 = (const float*)d_in[7];
  const float* b2  = (const float*)d_in[8];
  const float* Wt2 = (const float*)d_in[9];
  const float* bt2 = (const float*)d_in[10];
  const float* Wc2 = (const float*)d_in[11];
  const float* Wx3 = (const float*)d_in[12];
  const float* Wh3 = (const float*)d_in[13];
  const float* b3  = (const float*)d_in[14];
  const float* Wt3 = (const float*)d_in[15];
  const float* bt3 = (const float*)d_in[16];
  const float* Wc3 = (const float*)d_in[17];
  float* out = (float*)d_out;

  // ---- workspace carve-up ----
  char* p = (char*)d_ws;
  auto alloc = [&](size_t bytes) -> char* {
    char* r = p;
    p += (bytes + 255) & ~(size_t)255;
    return r;
  };
  __bf16* WxT[3];  // [4H][D] bf16
  __bf16* WhT[3];  // [4H][H] bf16
  for (int l = 0; l < 3; ++l) WxT[l] = (__bf16*)alloc((size_t)Gz * Dz * 2);
  for (int l = 0; l < 3; ++l) WhT[l] = (__bf16*)alloc((size_t)Gz * Hz * 2);
  __bf16* WtT2 = (__bf16*)alloc((size_t)Hz * 2 * Hz * 2);  // [H][2H]
  __bf16* WtT3 = (__bf16*)alloc((size_t)Hz * 2 * Hz * 2);
  __bf16* WcT2 = (__bf16*)alloc((size_t)Hz * Hz * 2);      // [H][H]
  __bf16* WcT3 = (__bf16*)alloc((size_t)Hz * Hz * 2);
  __bf16* ABF0 = (__bf16*)alloc((size_t)BTz * Hz * 2);
  __bf16* ABF1 = (__bf16*)alloc((size_t)BTz * Hz * 2);
  __bf16* CONCAT = (__bf16*)alloc((size_t)BTz * 2 * Hz * 2);
  float* XG = (float*)alloc((size_t)BTz * Gz * 4);
  float* F0 = (float*)alloc((size_t)BTz * Hz * 4);
  float* F1 = (float*)alloc((size_t)BTz * Hz * 4);
  float* F2 = (float*)alloc((size_t)BTz * Hz * 4);
  float* F3 = (float*)alloc((size_t)BTz * Hz * 4);
  __bf16* HBUF = (__bf16*)alloc((size_t)2 * Bz * Hz * 2);
  float* CST = (float*)alloc((size_t)Bz * Hz * 4);
  unsigned* CNT = (unsigned*)alloc(256);

  const int EW = 256;
  const int nBTH = BTz * Hz;
  const int gBTH = (nBTH + EW - 1) / EW;
  // one wave per 16x64 strip -> blocks of 8 waves
  auto gemm_blocks = [](int M, int N) { return ((M >> 4) * (N >> 6) * 32 + 255) / 256; };

  // ---- one-time (per launch) weight conversion: f32 [K,N] -> bf16 [N,K] ----
  {
    const int nW = Dz * Gz;
    const int gW = (nW + EW - 1) / EW;
    k_transpose_bf16<<<gW, EW, 0, stream>>>(Wx1, WxT[0], Dz, Gz);
    k_transpose_bf16<<<gW, EW, 0, stream>>>(Wx2, WxT[1], Hz, Gz);
    k_transpose_bf16<<<gW, EW, 0, stream>>>(Wx3, WxT[2], Hz, Gz);
    k_transpose_bf16<<<gW, EW, 0, stream>>>(Wh1, WhT[0], Hz, Gz);
    k_transpose_bf16<<<gW, EW, 0, stream>>>(Wh2, WhT[1], Hz, Gz);
    k_transpose_bf16<<<gW, EW, 0, stream>>>(Wh3, WhT[2], Hz, Gz);
    const int nT = 2 * Hz * Hz;
    const int gT = (nT + EW - 1) / EW;
    k_transpose_bf16<<<gT, EW, 0, stream>>>(Wt2, WtT2, 2 * Hz, Hz);
    k_transpose_bf16<<<gT, EW, 0, stream>>>(Wt3, WtT3, 2 * Hz, Hz);
    const int nC = Hz * Hz;
    const int gC = (nC + EW - 1) / EW;
    k_transpose_bf16<<<gC, EW, 0, stream>>>(Wc2, WcT2, Hz, Hz);
    k_transpose_bf16<<<gC, EW, 0, stream>>>(Wc3, WcT3, Hz, Hz);
  }

  // ================= Layer 1: forward LSTM (no highway) =================
  k_f32_to_bf16<<<gBTH, EW, 0, stream>>>(inputs, ABF0, nBTH);
  k_gemm4<<<gemm_blocks(BTz, Gz), 256, 0, stream>>>(ABF0, WxT[0], b1, XG, BTz, Gz, Dz);
  k_scan_init<<<(Bz * Hz + EW - 1) / EW, EW, 0, stream>>>(CNT, CST, HBUF);
  k_lstm_scan<<<NBLK, 128, 0, stream>>>(XG, WhT[0], length, F0, HBUF, CST, CNT);
  // F0 = y1

  // ================= Layer 2: reversed LSTM + highway =================
  k_reverse_bf16<<<gBTH, EW, 0, stream>>>(F0, length, ABF1);   // reversed input, bf16
  k_f32_to_bf16<<<gBTH, EW, 0, stream>>>(F0, ABF0, nBTH);      // y1 bf16 (for Wc)
  k_gemm4<<<gemm_blocks(BTz, Gz), 256, 0, stream>>>(ABF1, WxT[1], b2, XG, BTz, Gz, Hz);
  k_scan_init<<<(Bz * Hz + EW - 1) / EW, EW, 0, stream>>>(CNT, CST, HBUF);
  k_lstm_scan<<<NBLK, 128, 0, stream>>>(XG, WhT[1], length, F1, HBUF, CST, CNT);
  k_reverse_f32<<<gBTH, EW, 0, stream>>>(F1, length, F2);      // F2 = y2
  k_concat<<<gBTH, EW, 0, stream>>>(F2, F0, CONCAT);           // [prev_h(y2), x(y1)]
  k_gemm4<<<gemm_blocks(BTz, Hz), 256, 0, stream>>>(CONCAT, WtT2, bt2, F1, BTz, Hz, 2 * Hz);
  k_gemm4<<<gemm_blocks(BTz, Hz), 256, 0, stream>>>(ABF0, WcT2, nullptr, F3, BTz, Hz, Hz);
  k_combine<<<gBTH, EW, 0, stream>>>(F2, F1, F3, F0, nBTH);    // F0 = out2

  // ================= Layer 3: forward LSTM + highway =================
  k_f32_to_bf16<<<gBTH, EW, 0, stream>>>(F0, ABF0, nBTH);      // x3 bf16
  k_gemm4<<<gemm_blocks(BTz, Gz), 256, 0, stream>>>(ABF0, WxT[2], b3, XG, BTz, Gz, Hz);
  k_scan_init<<<(Bz * Hz + EW - 1) / EW, EW, 0, stream>>>(CNT, CST, HBUF);
  k_lstm_scan<<<NBLK, 128, 0, stream>>>(XG, WhT[2], length, F1, HBUF, CST, CNT);
  k_concat<<<gBTH, EW, 0, stream>>>(F1, F0, CONCAT);           // [prev_h(y3), x3]
  k_gemm4<<<gemm_blocks(BTz, Hz), 256, 0, stream>>>(CONCAT, WtT3, bt3, F2, BTz, Hz, 2 * Hz);
  k_gemm4<<<gemm_blocks(BTz, Hz), 256, 0, stream>>>(ABF0, WcT3, nullptr, F3, BTz, Hz, Hz);
  k_combine<<<gBTH, EW, 0, stream>>>(F1, F2, F3, out, nBTH);   // final output
}
